// AttentionMix_73718818669265
// MI455X (gfx1250) — matrix operand order repaired and checked by reading, not compiled, not run on hardware
//
#include <hip/hip_runtime.h>

typedef __attribute__((ext_vector_type(16))) _Float16 v16h;
typedef __attribute__((ext_vector_type(8)))  float    v8f;

#define NUM_HEADS 2
#define C_IN      64
#define P_TOT     2048
#define N_NEI     32
#define TP        16      // p-points per main-kernel block
#define KV_STRIDE 36      // V scratch row stride (floats): mult of 4, conflict-free
#define KT_STRIDE 68      // transposed-K scratch row stride (floats): mult of 4
#define QSTRIDE   132     // s_q row stride (floats), 16B-aligned rows

// B-fragment K mapping (dense 32x16 B: lanes 0-15 <-> K 0..15, lanes 16-31 <-> K 16..31)
__device__ __forceinline__ int bOff(int j, int g) {
    return j + 16 * g;
}

// ---------------------------------------------------------------------------
// Pre-pass: yT[b][p][n][c] (f16, c-contiguous) from y[b][c][p][n] (f32).
// One block per (b,p). Coalesced reads, LDS-tiled transpose, f16x2 writes.
// ---------------------------------------------------------------------------
__global__ __launch_bounds__(256, 4)
void transpose_y_kernel(const float* __restrict__ y, _Float16* __restrict__ yT)
{
    __shared__ float tile[C_IN * 33];      // 64 x 32, pad 33 -> conflict-free
    const int tid = threadIdx.x;
    const int b   = blockIdx.x / P_TOT;
    const int p   = blockIdx.x % P_TOT;

    for (int it = 0; it < 8; ++it) {
        int e = it * 256 + tid;            // 0..2047
        int c = e >> 5, n = e & 31;
        tile[c * 33 + n] = y[(((size_t)b * C_IN + c) * P_TOT + p) * N_NEI + n];
    }
    __syncthreads();

    _Float16* dst = yT + ((size_t)(b * P_TOT + p)) * (N_NEI * C_IN);
    const int n0 = tid >> 5;               // 0..7
    const int c0 = (tid & 31) * 2;         // even c
    for (int it = 0; it < 4; ++it) {
        int n = n0 + it * 8;
        _Float16 lo = (_Float16)tile[(c0 + 0) * 33 + n];
        _Float16 hi = (_Float16)tile[(c0 + 1) * 33 + n];
        union { _Float16 h[2]; unsigned int u; } pk;
        pk.h[0] = lo; pk.h[1] = hi;
        *(unsigned int*)&dst[n * C_IN + c0] = pk.u;   // 128B contiguous per wave
    }
}

// ---------------------------------------------------------------------------
// Main fused kernel
// ---------------------------------------------------------------------------
__global__ __launch_bounds__(256, 1)
void attn_mix_kernel(const float* __restrict__ x,
                     const _Float16* __restrict__ yT,
                     const unsigned char* __restrict__ mask,
                     const float* __restrict__ qw,
                     const float* __restrict__ kw,
                     const float* __restrict__ vw,
                     const float* __restrict__ ow,
                     float* __restrict__ out)
{
    // ---- LDS ----
    __shared__ _Float16 s_qw[8192];            // 128x64, swizzled A tiles (8 mt x 2 kc) * 512
    __shared__ _Float16 s_kw[8192];
    __shared__ _Float16 s_vw[8192];
    __shared__ _Float16 s_ow[8192];            // 64x128, swizzled A tiles (4 mt x 4 kc) * 512
    __shared__ float    s_q[TP * QSTRIDE];     // q * (1/sqrt(C)); [pl][row]
    __shared__ _Float16 s_att[128 * TP];       // attention output, [c][pl] f16
    __shared__ float    s_kv[8][C_IN * KV_STRIDE]; // per-wave scratch: kT (32x68) then V (64x36)
    __shared__ float    s_w[8][N_NEI];         // per-wave softmax weights

    const int tid  = threadIdx.x;
    const int lane = tid & 31;
    const int wave = tid >> 5;
    const int g    = (lane >> 4) & 1;
    const int ln   = lane & 15;

    const int blk = blockIdx.x;                // 512 blocks
    const int b   = blk / (P_TOT / TP);
    const int p0  = (blk % (P_TOT / TP)) * TP;

    // ---- Phase A: stage weights to LDS as f16, pre-swizzled into A-fragment layout ----
    for (int idx = tid; idx < 128 * 64; idx += 256) {      // q_w / k_w / v_w : 128x64
        int row = idx >> 6, col = idx & 63;
        int mt = row >> 4, m = row & 15;
        int kc = col >> 5, kp = col & 31;
        int gg = (kp >> 4) & 1;
        int j  = kp & 15;                                  // contiguous A K-map (k = 16*half + j)
        int dst = ((mt * 2 + kc) * 512) + (gg * 16 + m) * 16 + j;
        s_qw[dst] = (_Float16)qw[idx];
        s_kw[dst] = (_Float16)kw[idx];
        s_vw[dst] = (_Float16)vw[idx];
    }
    for (int idx = tid; idx < 64 * 128; idx += 256) {      // out_w : 64x128
        int row = idx >> 7, col = idx & 127;
        int mt = row >> 4, m = row & 15;
        int kc = col >> 5, kp = col & 31;
        int gg = (kp >> 4) & 1;
        int j  = kp & 15;
        int dst = ((mt * 4 + kc) * 512) + (gg * 16 + m) * 16 + j;
        s_ow[dst] = (_Float16)ow[idx];
    }
    __syncthreads();

    // ---- Phase Q: Q = q_w(128x64) @ x_tile(64x16); wave w computes M-tile w ----
    {
        v16h bx[2];
        for (int kc = 0; kc < 2; ++kc)
            for (int j = 0; j < 16; ++j) {
                int c = bOff(j, g) + 32 * kc;
                bx[kc][j] = (_Float16)x[((size_t)b * C_IN + c) * P_TOT + p0 + ln];
            }
        v8f acc = {};
        for (int kc = 0; kc < 2; ++kc) {
            v16h a = *(const v16h*)&s_qw[(wave * 2 + kc) * 512 + lane * 16];
            acc = __builtin_amdgcn_wmma_f32_16x16x32_f16(false, a, false, bx[kc],
                                                         (short)0, acc, false, false);
        }
        for (int r = 0; r < 8; ++r) {
            int m = 16 * wave + r + 8 * g;
            s_q[ln * QSTRIDE + m] = acc[r] * 0.125f;   // fold 1/sqrt(64)
        }
    }
    __syncthreads();

    // ---- Phase B: per-point attention; wave handles pl = 2*wave + {0,1} ----
    for (int pp = 0; pp < 2; ++pp) {
        const int pl = wave * 2 + pp;
        const int p  = p0 + pl;

        // y B-fragments for this point: single 32B aligned loads from c-contiguous yT
        const _Float16* yt = yT + ((size_t)(b * P_TOT + p)) * (N_NEI * C_IN);
        v16h by[2][2];
        for (int kc = 0; kc < 2; ++kc)
            for (int nt = 0; nt < 2; ++nt)
                by[kc][nt] = *(const v16h*)&yt[(ln + 16 * nt) * C_IN + 16 * g + 32 * kc];

        const bool msk = mask[((size_t)b * P_TOT + p) * N_NEI + lane];
        float* kv = s_kv[wave];

        for (int h = 0; h < NUM_HEADS; ++h) {
            // K-projection, stored TRANSPOSED: kT[n][c], stride 68.
            // Consecutive r are consecutive floats -> stores still merge to b128.
            for (int mt = 0; mt < 4; ++mt) {
                int mtg = h * 4 + mt;
                v16h a0 = *(const v16h*)&s_kw[(mtg * 2 + 0) * 512 + lane * 16];
                v16h a1 = *(const v16h*)&s_kw[(mtg * 2 + 1) * 512 + lane * 16];
                for (int nt = 0; nt < 2; ++nt) {
                    v8f acc = {};
                    acc = __builtin_amdgcn_wmma_f32_16x16x32_f16(false, a0, false, by[0][nt],
                                                                 (short)0, acc, false, false);
                    acc = __builtin_amdgcn_wmma_f32_16x16x32_f16(false, a1, false, by[1][nt],
                                                                 (short)0, acc, false, false);
                    for (int r = 0; r < 8; ++r)
                        kv[(ln + 16 * nt) * KT_STRIDE + mt * 16 + r + 8 * g] = acc[r];
                }
            }

            // attention logits: lane <-> neighbor n; both operands as float4
            float wv = 0.f;
            for (int c = 0; c < C_IN; c += 4) {
                float4 q4 = *(const float4*)&s_q[pl * QSTRIDE + h * 64 + c];
                float4 k4 = *(const float4*)&kv[lane * KT_STRIDE + c];
                wv += q4.x * k4.x + q4.y * k4.y + q4.z * k4.z + q4.w * k4.w;
            }
            if (msk) wv = -1e9f;

            // softmax across 32 lanes (wave32)
            float mx = wv;
            for (int off = 16; off > 0; off >>= 1)
                mx = fmaxf(mx, __shfl_xor(mx, off, 32));
            float e = __expf(wv - mx);
            float sm = e;
            for (int off = 16; off > 0; off >>= 1)
                sm += __shfl_xor(sm, off, 32);
            s_w[wave][lane] = e / sm;

            // V-projection, row-major v[c][n] (stride 36), reuses same scratch
            for (int mt = 0; mt < 4; ++mt) {
                int mtg = h * 4 + mt;
                v16h a0 = *(const v16h*)&s_vw[(mtg * 2 + 0) * 512 + lane * 16];
                v16h a1 = *(const v16h*)&s_vw[(mtg * 2 + 1) * 512 + lane * 16];
                for (int nt = 0; nt < 2; ++nt) {
                    v8f acc = {};
                    acc = __builtin_amdgcn_wmma_f32_16x16x32_f16(false, a0, false, by[0][nt],
                                                                 (short)0, acc, false, false);
                    acc = __builtin_amdgcn_wmma_f32_16x16x32_f16(false, a1, false, by[1][nt],
                                                                 (short)0, acc, false, false);
                    for (int r = 0; r < 8; ++r)
                        kv[(mt * 16 + r + 8 * g) * KV_STRIDE + ln + 16 * nt] = acc[r];
                }
            }

            // out[c] = sum_n w[n] * v[c][n]; all-float4 DS traffic
            for (int it = 0; it < 2; ++it) {
                int c = it * 32 + lane;
                float o = 0.f;
                for (int n = 0; n < N_NEI; n += 4) {
                    float4 w4 = *(const float4*)&s_w[wave][n];
                    float4 v4 = *(const float4*)&kv[c * KV_STRIDE + n];
                    o += w4.x * v4.x + w4.y * v4.y + w4.z * v4.z + w4.w * v4.w;
                }
                s_att[(h * 64 + c) * TP + pl] = (_Float16)o;
            }
        }
    }
    __syncthreads();

    // ---- Phase C: result = out_w(64x128) @ att(128x16) + x ; waves 0..3 ----
    if (wave < 4) {
        const int mt = wave;
        v8f acc = {};
        for (int kc = 0; kc < 4; ++kc) {
            v16h bf;
            for (int j = 0; j < 16; ++j) {
                int c = bOff(j, g) + 32 * kc;
                bf[j] = s_att[c * TP + ln];
            }
            v16h a = *(const v16h*)&s_ow[(mt * 4 + kc) * 512 + lane * 16];
            acc = __builtin_amdgcn_wmma_f32_16x16x32_f16(false, a, false, bf,
                                                         (short)0, acc, false, false);
        }
        for (int r = 0; r < 8; ++r) {
            int o = mt * 16 + r + 8 * g;
            size_t gidx = ((size_t)b * C_IN + o) * P_TOT + p0 + ln;
            out[gidx] = acc[r] + x[gidx];
        }
    }
}

extern "C" void kernel_launch(void* const* d_in, const int* in_sizes, int n_in,
                              void* d_out, int out_size, void* d_ws, size_t ws_size,
                              hipStream_t stream) {
    const float*         x    = (const float*)d_in[0];
    const float*         y    = (const float*)d_in[1];
    const unsigned char* mask = (const unsigned char*)d_in[2];   // jax bool -> 1 byte
    const float*         qw   = (const float*)d_in[3];
    const float*         kw   = (const float*)d_in[4];
    const float*         vw   = (const float*)d_in[5];
    const float*         ow   = (const float*)d_in[6];
    float*               out  = (float*)d_out;

    _Float16* yT = (_Float16*)d_ws;   // 4*2048*32*64 f16 = 32 MiB scratch

    transpose_y_kernel<<<4 * P_TOT, 256, 0, stream>>>(y, yT);

    const int blocks = 4 * (P_TOT / TP);   // 512
    attn_mix_kernel<<<blocks, 256, 0, stream>>>(x, yT, mask, qw, kw, vw, ow, out);
}
